// nlblock_16174846837015
// MI455X (gfx1250) — matrix operand order repaired and checked by reading, not compiled, not run on hardware
//
#include <hip/hip_runtime.h>
#include <hip/hip_bf16.h>

// ---------------------------------------------------------------------------
// Non-local block (attention) for MI455X / gfx1250, wave32 + WMMA f16.
// B=4, C=256, H=W=64, N=4096, head dim Cp=128.
// ---------------------------------------------------------------------------

typedef _Float16 v16h __attribute__((ext_vector_type(16)));
typedef float    v8f  __attribute__((ext_vector_type(8)));
typedef float    f32x4 __attribute__((ext_vector_type(4)));

union U16 { v16h h; f32x4 f[2]; };
union H8  { f32x4 f; _Float16 h[8]; };

#define NB   4
#define NC   256
#define NCP  128
#define NN   4096   // H*W

__device__ __forceinline__ v8f wmma_f16(v16h a, v16h b, v8f c) {
  // v_wmma_f32_16x16x32_f16  D = A*B + C   (wave32)
  return __builtin_amdgcn_wmma_f32_16x16x32_f16(
      /*neg_a=*/false, a, /*neg_b=*/false, b,
      /*c_mod=*/(short)0, c, /*reuse_a=*/false, /*reuse_b=*/false);
}

// A-fragment (16x32 f16, MxK): lane holds row M=lane&15.
// halfs[0..7]  = K (8*hi .. 8*hi+7), halfs[8..15] = K (16+8*hi .. 23+8*hi)
__device__ __forceinline__ v16h load_frag_a(const _Float16* row, int hi) {
  U16 u;
  u.f[0] = *(const f32x4*)(row + 8 * hi);
  u.f[1] = *(const f32x4*)(row + 16 + 8 * hi);
  return u.h;
}

// B-fragment (32x16 f16, KxN): lane holds col N=lane&15.
// halfs[0..7] = K (8*hi .. 8*hi+7), halfs[8..15] = K (16+8*hi .. 23+8*hi), memory [col][k].
__device__ __forceinline__ v16h load_frag_b(const _Float16* colrow, int hi) {
  const _Float16* q = colrow + 8 * hi;
  U16 u;
  u.f[0] = *(const f32x4*)(q);
  u.f[1] = *(const f32x4*)(q + 16);
  return u.h;
}

__device__ __forceinline__ v8f zero8() {
  v8f z = {0.f, 0.f, 0.f, 0.f, 0.f, 0.f, 0.f, 0.f};
  return z;
}

__device__ __forceinline__ void wave_lds_fence() {
  // wave-private LDS producer->consumer: wait for own ds stores.
  asm volatile("s_wait_dscnt 0" ::: "memory");
}

// ---------------------------------------------------------------------------
// 1) Convert the four weight matrices to f16 (contiguous: wth|wpi|wg|wz).
// ---------------------------------------------------------------------------
__global__ void k_convert_w(const float* __restrict__ wt, const float* __restrict__ wp,
                            const float* __restrict__ wg, const float* __restrict__ wz,
                            _Float16* __restrict__ out) {
  int i = blockIdx.x * blockDim.x + threadIdx.x;          // 0 .. 131071
  int a = i >> 15;                                        // which matrix
  int r = i & 32767;
  const float* src = (a == 0) ? wt : (a == 1) ? wp : (a == 2) ? wg : wz;
  out[i] = (_Float16)src[r];
}

// ---------------------------------------------------------------------------
// 2) xp = x + pos : f32 [b][c][n] (residual) and f16 transposed [b][n][c].
// ---------------------------------------------------------------------------
__global__ void k_prep(const float* __restrict__ x, const float* __restrict__ pos,
                       float* __restrict__ xp, _Float16* __restrict__ xph) {
  int bid = blockIdx.x;                 // b(4) * cb(8) * nb(128)
  int nb = bid & 127;
  int t  = bid >> 7;
  int cb = t & 7;
  int b  = t >> 3;
  int c0 = cb * 32, n0 = nb * 32;

  __shared__ float tile[32][33];
  int j  = threadIdx.x & 31;
  int i0 = threadIdx.x >> 5;            // 0..7
#pragma unroll
  for (int rep = 0; rep < 4; ++rep) {
    int i = i0 + rep * 8;
    size_t src = ((size_t)(b * NC + c0 + i)) * NN + n0 + j;
    float v = x[src] + pos[((size_t)(c0 + i)) * NN + n0 + j];
    xp[src] = v;
    tile[i][j] = v;
  }
  __syncthreads();
#pragma unroll
  for (int rep = 0; rep < 4; ++rep) {
    int jj = i0 + rep * 8;
    int ii = j;
    xph[((size_t)b * NN + n0 + jj) * NC + c0 + ii] = (_Float16)tile[ii][jj];
  }
}

// ---------------------------------------------------------------------------
// 3) Projections: theta/pi -> [b][n][c] f16 (n-major), g -> [b][c][n] f16.
//    GEMM: A = W (128x256), B = xp (256 x N) read from xph[b][n][c].
//    grid = B * (N/64), block = 128 (4 waves).
// ---------------------------------------------------------------------------
__global__ void __launch_bounds__(128) k_proj(
    const _Float16* __restrict__ wth, const _Float16* __restrict__ wpi,
    const _Float16* __restrict__ wg,  const _Float16* __restrict__ xph,
    _Float16* __restrict__ thetaT, _Float16* __restrict__ piT,
    _Float16* __restrict__ gcm) {
  int b = blockIdx.x >> 6;
  int n0 = (blockIdx.x & 63) * 64;
  int w = threadIdx.x >> 5, lane = threadIdx.x & 31;
  int li = lane & 15, hi = lane >> 4;

  for (int mt = 0; mt < 3; ++mt) {
    const _Float16* W = (mt == 0) ? wth : (mt == 1) ? wpi : wg;
#pragma unroll
    for (int mr = 0; mr < 2; ++mr) {
      int obase = (w * 2 + mr) * 16;
      const _Float16* arow = W + (size_t)(obase + li) * NC;
#pragma unroll
      for (int nc = 0; nc < 4; ++nc) {
        int ncol = n0 + nc * 16 + li;
        const _Float16* brow = xph + ((size_t)b * NN + ncol) * NC;
        v8f acc = zero8();
#pragma unroll
        for (int k = 0; k < 8; ++k) {       // K = 256 = 8 * 32
          v16h A = load_frag_a(arow + k * 32, hi);
          v16h Bf = load_frag_b(brow + k * 32, hi);
          acc = wmma_f16(A, Bf, acc);
        }
        if (mt < 2) {
          // D rows = o (r + 8*hi), col = ncol. n-major store: 8 contiguous
          // o-values per lane -> one b128 store.
          H8 pk;
#pragma unroll
          for (int r = 0; r < 8; ++r) pk.h[r] = (_Float16)acc[r];
          _Float16* dst = ((mt == 0) ? thetaT : piT) +
                          ((size_t)b * NN + ncol) * NCP + obase + 8 * hi;
          *(f32x4*)dst = pk.f;
        } else {
          // g stored c-major [b][o][n]
#pragma unroll
          for (int r = 0; r < 8; ++r) {
            int o = obase + r + 8 * hi;
            gcm[((size_t)b * NCP + o) * NN + ncol] = (_Float16)acc[r];
          }
        }
      }
    }
  }
}

// ---------------------------------------------------------------------------
// 4) Fused flash attention: S = theta^T pi, online softmax, O = P g^T.
//    grid = B * (N/64) query blocks, block = 128 (4 waves x 16 query rows).
//    Output O stored f16 n-major: out_nm[b][n][c].
// ---------------------------------------------------------------------------
__global__ void __launch_bounds__(128) k_attn(
    const _Float16* __restrict__ thetaT, const _Float16* __restrict__ piT,
    const _Float16* __restrict__ gcm, _Float16* __restrict__ out_nm) {
  int b = blockIdx.x >> 6;
  int qb = blockIdx.x & 63;
  int w = threadIdx.x >> 5, lane = threadIdx.x & 31;
  int li = lane & 15, hi = lane >> 4;
  int nw = qb * 64 + w * 16;                       // this wave's 16 query rows

  __shared__ _Float16 ldsP[4][16][64];             // per-wave P tile (8KB total)

  const _Float16* arow = thetaT + ((size_t)b * NN + nw + li) * NCP;

  v8f accO[8];
  float mrun[8], lrun[8];
#pragma unroll
  for (int i = 0; i < 8; ++i) { accO[i] = zero8(); mrun[i] = -3.0e38f; lrun[i] = 0.f; }

  for (int m0 = 0; m0 < NN; m0 += 64) {
    if (m0 + 64 < NN)
      __builtin_prefetch(piT + ((size_t)b * NN + m0 + 64 + li) * NCP, 0, 1);

    // ---- S tiles: 16 queries x 64 keys, K = 128 channels -------------------
    v8f S[4] = {zero8(), zero8(), zero8(), zero8()};
#pragma unroll
    for (int k = 0; k < 4; ++k) {                  // K = 128 = 4 * 32
      v16h A = load_frag_a(arow + k * 32, hi);
#pragma unroll
      for (int j = 0; j < 4; ++j) {
        const _Float16* brow =
            piT + ((size_t)b * NN + m0 + j * 16 + li) * NCP + k * 32;
        S[j] = wmma_f16(A, load_frag_b(brow, hi), S[j]);
      }
    }

    // ---- online softmax (row = r + 8*hi lives in a 16-lane group) ---------
    float scale[8];
#pragma unroll
    for (int r = 0; r < 8; ++r) {
      float mx = fmaxf(fmaxf(S[0][r], S[1][r]), fmaxf(S[2][r], S[3][r]));
#pragma unroll
      for (int msk = 1; msk < 16; msk <<= 1) mx = fmaxf(mx, __shfl_xor(mx, msk, 32));
      float nm = fmaxf(mrun[r], mx);
      float sc = __expf(mrun[r] - nm);
      float rs = 0.f;
#pragma unroll
      for (int j = 0; j < 4; ++j) {
        float p = __expf(S[j][r] - nm);
        S[j][r] = p;
        rs += p;
      }
#pragma unroll
      for (int msk = 1; msk < 16; msk <<= 1) rs += __shfl_xor(rs, msk, 32);
      lrun[r] = lrun[r] * sc + rs;
      mrun[r] = nm;
      scale[r] = sc;
    }
#pragma unroll
    for (int ct = 0; ct < 8; ++ct)
#pragma unroll
      for (int r = 0; r < 8; ++r) accO[ct][r] *= scale[r];

    // ---- P (D-layout f32) -> LDS f16, then re-read in A-fragment layout ---
#pragma unroll
    for (int j = 0; j < 4; ++j)
#pragma unroll
      for (int r = 0; r < 8; ++r)
        ldsP[w][r + 8 * hi][j * 16 + li] = (_Float16)S[j][r];
    wave_lds_fence();

    // ---- O += P @ g^T : A = P (16 x 64), B = g_cm (K=m x col=c) ------------
#pragma unroll
    for (int kk = 0; kk < 2; ++kk) {               // K = 64 keys = 2 * 32
      v16h A = load_frag_a(&ldsP[w][li][kk * 32], hi);
#pragma unroll
      for (int ct = 0; ct < 8; ++ct) {
        const _Float16* brow =
            gcm + ((size_t)b * NCP + ct * 16 + li) * NN + m0 + kk * 32;
        accO[ct] = wmma_f16(A, load_frag_b(brow, hi), accO[ct]);
      }
    }
  }

  // ---- epilogue: divide by row sums, store n-major f16 ---------------------
  float invl[8];
#pragma unroll
  for (int r = 0; r < 8; ++r) invl[r] = 1.0f / lrun[r];
#pragma unroll
  for (int ct = 0; ct < 8; ++ct) {
    int c = ct * 16 + li;
#pragma unroll
    for (int r = 0; r < 8; ++r) {
      int n = nw + r + 8 * hi;
      out_nm[((size_t)b * NN + n) * NCP + c] = (_Float16)(accO[ct][r] * invl[r]);
    }
  }
}

// ---------------------------------------------------------------------------
// 5) z = w_z @ O^T : A = wz (256x128), B = out_nm[b][n][c]. Also accumulate
//    per-channel sum / sumsq for BatchNorm. grid = B*(N/64), block = 128.
// ---------------------------------------------------------------------------
__global__ void __launch_bounds__(128) k_zgemm(
    const _Float16* __restrict__ wz, const _Float16* __restrict__ out_nm,
    float* __restrict__ z, float* __restrict__ bnsum, float* __restrict__ bnsq) {
  int b = blockIdx.x >> 6;
  int n0 = (blockIdx.x & 63) * 64;
  int w = threadIdx.x >> 5, lane = threadIdx.x & 31;
  int li = lane & 15, hi = lane >> 4;

#pragma unroll
  for (int ot = 0; ot < 4; ++ot) {
    int obase = (w * 4 + ot) * 16;
    const _Float16* arow = wz + (size_t)(obase + li) * NCP;
#pragma unroll
    for (int nc = 0; nc < 4; ++nc) {
      int ncol = n0 + nc * 16 + li;
      const _Float16* brow = out_nm + ((size_t)b * NN + ncol) * NCP;
      v8f acc = zero8();
#pragma unroll
      for (int k = 0; k < 4; ++k)                  // K = 128 = 4 * 32
        acc = wmma_f16(load_frag_a(arow + k * 32, hi),
                       load_frag_b(brow + k * 32, hi), acc);
#pragma unroll
      for (int r = 0; r < 8; ++r) {
        int o = obase + r + 8 * hi;
        float v = acc[r];
        z[((size_t)b * NC + o) * NN + ncol] = v;
        float s = v, q = v * v;
#pragma unroll
        for (int msk = 1; msk < 16; msk <<= 1) {
          s += __shfl_xor(s, msk, 32);
          q += __shfl_xor(q, msk, 32);
        }
        if (li == 0) {
          atomicAdd(&bnsum[o], s);
          atomicAdd(&bnsq[o], q);
        }
      }
    }
  }
}

// ---------------------------------------------------------------------------
// 6) BN statistics and fused residual+affine finalize.
// ---------------------------------------------------------------------------
__global__ void k_zero(float* __restrict__ p) {
  p[blockIdx.x * blockDim.x + threadIdx.x] = 0.f;
}

__global__ void k_bnstats(const float* __restrict__ s, const float* __restrict__ q,
                          float* __restrict__ mean, float* __restrict__ inv) {
  int c = threadIdx.x;
  const float cnt = (float)(NB * NN);
  float m = s[c] / cnt;
  float v = q[c] / cnt - m * m;
  mean[c] = m;
  inv[c] = rsqrtf(v + 1e-5f);
}

__global__ void k_final(const float* __restrict__ xp, const float* __restrict__ z,
                        const float* __restrict__ mean, const float* __restrict__ inv,
                        const float* __restrict__ gamma, const float* __restrict__ beta,
                        float* __restrict__ out) {
  size_t i = ((size_t)blockIdx.x * blockDim.x + threadIdx.x) * 4;
  int c = (int)((i >> 12) & (NC - 1));             // (i / 4096) % 256
  float a = inv[c] * gamma[c];
  float bb = beta[c] - mean[c] * a;
  f32x4 xv = *(const f32x4*)(xp + i);
  f32x4 zv = *(const f32x4*)(z + i);
  f32x4 o = xv + zv * a + bb;
  *(f32x4*)(out + i) = o;
}

// ---------------------------------------------------------------------------
// Launch
// ---------------------------------------------------------------------------
extern "C" void kernel_launch(void* const* d_in, const int* in_sizes, int n_in,
                              void* d_out, int out_size, void* d_ws, size_t ws_size,
                              hipStream_t stream) {
  const float* x       = (const float*)d_in[0];
  const float* w_theta = (const float*)d_in[1];
  const float* w_pi    = (const float*)d_in[2];
  const float* w_g     = (const float*)d_in[3];
  const float* w_z     = (const float*)d_in[4];
  const float* gamma   = (const float*)d_in[5];
  const float* beta    = (const float*)d_in[6];
  const float* pos     = (const float*)d_in[7];
  float* out = (float*)d_out;

  char* ws = (char*)d_ws;
  // f16 weights: [theta | pi | g | z], 32768 halfs each
  _Float16* wth_h  = (_Float16*)(ws + 0);
  _Float16* wpi_h  = (_Float16*)(ws + 65536);
  _Float16* wg_h   = (_Float16*)(ws + 131072);
  _Float16* wz_h   = (_Float16*)(ws + 196608);
  float*    bnsum  = (float*)(ws + 262144);        // 256 f32
  float*    bnsq   = (float*)(ws + 263168);        // 256 f32 (contiguous)
  float*    meanp  = (float*)(ws + 264192);
  float*    invp   = (float*)(ws + 265216);
  float*    xp     = (float*)(ws + 266240);        // B*C*N f32   (16 MB)
  _Float16* xph    = (_Float16*)(ws + 17043456);   // B*N*C f16   ( 8 MB)
  _Float16* thetaT = (_Float16*)(ws + 25432064);   // B*N*Cp f16  ( 4 MB)
  _Float16* piT    = (_Float16*)(ws + 29626368);   // B*N*Cp f16  ( 4 MB)
  _Float16* gcm    = (_Float16*)(ws + 33820672);   // B*Cp*N f16  ( 4 MB)
  _Float16* out_nm = (_Float16*)(ws + 38014976);   // B*N*Cp f16  ( 4 MB)
  float*    zbuf   = (float*)(ws + 42209280);      // B*C*N f32   (16 MB)
  (void)in_sizes; (void)n_in; (void)out_size; (void)ws_size;

  k_convert_w<<<512, 256, 0, stream>>>(w_theta, w_pi, w_g, w_z, wth_h);
  k_prep<<<NB * 8 * 128, 256, 0, stream>>>(x, pos, xp, xph);
  k_proj<<<NB * (NN / 64), 128, 0, stream>>>(wth_h, wpi_h, wg_h, xph,
                                             thetaT, piT, gcm);
  k_zero<<<2, 256, 0, stream>>>(bnsum);            // zeros bnsum + bnsq
  k_attn<<<NB * (NN / 64), 128, 0, stream>>>(thetaT, piT, gcm, out_nm);
  k_zgemm<<<NB * (NN / 64), 128, 0, stream>>>(wz_h, out_nm, zbuf, bnsum, bnsq);
  k_bnstats<<<1, 256, 0, stream>>>(bnsum, bnsq, meanp, invp);
  k_final<<<(NB * NC * NN) / (256 * 4), 256, 0, stream>>>(
      xp, zbuf, meanp, invp, gamma, beta, out);
}
